// DARTSModel_68831145885840
// MI455X (gfx1250) — matrix operand order repaired and checked by reading, not compiled, not run on hardware
//
#include <hip/hip_runtime.h>

// ================= types =================
typedef __attribute__((ext_vector_type(16))) __bf16 bf16x16;
typedef __attribute__((ext_vector_type(8)))  __bf16 bf16x8;
typedef __attribute__((ext_vector_type(8)))  float  f32x8;
typedef __attribute__((ext_vector_type(4)))  float  f32x4;

#define DEVI static __device__ __forceinline__

// ================= problem constants =================
#define NB   128      // batch
#define NS   256      // seq len
#define NE   512      // embed
#define NH   1024     // hidden
#define N2H  2048
#define NC   42

#define RNN_BLOCKS 64
#define RNN_THREADS 256

// ================= workspace layout (bytes) =================
#define OFF_BAR   ((size_t)0)                          // 2 ints barrier
#define OFF_AGG   ((size_t)256)                        // [128][1024] f32
#define OFF_ST    (OFF_AGG + (size_t)NB*NH*4)          // 10 states bf16 [128][1024]
#define ST_ELEMS  ((size_t)NB*NH)
#define ST_BYTES  (ST_ELEMS*2)
#define OFF_ENC   (OFF_ST + 10*ST_BYTES)               // [S*B][E] bf16 (row = s*128+b)
#define OFF_PENCW (OFF_ENC + (size_t)NS*NB*NE*2)       // packed enc_W  K=512 N=512
#define OFF_PW0X  (OFF_PENCW + (size_t)NE*NE*2)        // packed W0[:512]  K=512 N=2048
#define OFF_PW0H  (OFF_PW0X + (size_t)NE*N2H*2)        // packed W0[512:]  K=1024 N=2048
#define OFF_PWS   (OFF_PW0H + (size_t)NH*N2H*2)        // 8 packed Ws      K=1024 N=2048

#define N_ENCW ((size_t)NE*NE)
#define N_W0X  ((size_t)NE*N2H)
#define N_W0H  ((size_t)NH*N2H)
#define N_WSM  ((size_t)NH*N2H)
#define N_PACK_TOTAL (N_ENCW + N_W0X + N_W0H + 8*N_WSM)

// ================= WMMA helper =================
DEVI f32x8 wmma_bf16(bf16x16 a, bf16x16 b, f32x8 c) {
    // D = A(16x32) * B(32x16) + C, fp32 accumulate
    return __builtin_amdgcn_wmma_f32_16x16x32_bf16(
        /*neg_a*/false, a, /*neg_b*/false, b,
        /*c_mod*/(short)0, c, /*reuse_a*/false, /*reuse_b*/false);
}

union BF16x16u { bf16x16 v; bf16x8 h[2]; };

// A fragment from row-major bf16 matrix (ld elements per row).
// Lane l -> row m = mbase + (l&15); k-octets at kt*32 + (l>>4)*8 and +16.
DEVI bf16x16 loadAfrag(const __bf16* base, int ld, int mbase, int kt, int lane) {
    int m  = mbase + (lane & 15);
    int k0 = kt * 32 + ((lane >> 4) << 3);
    const __bf16* p = base + (size_t)m * ld + k0;
    BF16x16u r;
    r.h[0] = *(const bf16x8*)(p);
    r.h[1] = *(const bf16x8*)(p + 16);
    return r.v;
}

// B fragment from pre-packed layout: frag(nt,kt) is 32 lanes x 16 bf16 contiguous.
DEVI bf16x16 loadBfrag(const __bf16* pack, int nKt, int nt, int kt, int lane) {
    const __bf16* p = pack + ((((size_t)nt * nKt + kt) * 32) + lane) * 16;
    return *(const bf16x16*)p;
}

DEVI float sigmoidf_(float x) { return 1.0f / (1.0f + __expf(-x)); }

DEVI float apply_act(int act, float x) {
    switch (act) {
        case 0:  return tanhf(x);
        case 1:  return fmaxf(x, 0.0f);
        case 2:  return sigmoidf_(x);
        default: return x;
    }
}

// ---- async global -> LDS staging (CDNA5 path, ASYNCcnt-tracked) ----
// Copies `nbytes` from global `src` to LDS `dst` cooperatively (16B/chunk).
DEVI void stage_panel_async(const __bf16* src, __bf16* dst, int nbytes, int tid) {
    for (int ofs = tid * 16; ofs < nbytes; ofs += RNN_THREADS * 16) {
        unsigned ldsoff = (unsigned)(size_t)((const char*)dst + ofs); // low 32b = LDS offset
        const void* g = (const char*)src + ofs;
        asm volatile("global_load_async_to_lds_b128 %0, %1, off"
                     :: "v"(ldsoff), "v"(g) : "memory");
    }
    asm volatile("s_wait_asynccnt 0" ::: "memory");
}

// GEMM accumulate: 4 M-tiles (rows 0..63 of the staged LDS panel) x paired
// N-tiles (cols pair*16.. and 1024+pair*16..). A from LDS, B packed global.
DEVI void gemm_lds(const __bf16* Alds, int lda,
                   const __bf16* Bpack, int K, int pair,
                   f32x8 accC[4], f32x8 accH[4], int lane) {
    int nKt = K >> 5;
    for (int kt = 0; kt < nKt; ++kt) {
        if (kt + 1 < nKt) { // gentle prefetch of next weight fragments (L2 resident)
            __builtin_prefetch(Bpack + ((((size_t)pair      * nKt + kt + 1) * 32) + lane) * 16, 0, 0);
            __builtin_prefetch(Bpack + ((((size_t)(pair+64) * nKt + kt + 1) * 32) + lane) * 16, 0, 0);
        }
        bf16x16 bc = loadBfrag(Bpack, nKt, pair,      kt, lane);
        bf16x16 bh = loadBfrag(Bpack, nKt, pair + 64, kt, lane);
#pragma unroll
        for (int i = 0; i < 4; ++i) {
            bf16x16 a = loadAfrag(Alds, lda, i * 16, kt, lane);
            accC[i] = wmma_bf16(a, bc, accC[i]);
            accH[i] = wmma_bf16(a, bh, accH[i]);
        }
    }
}

// Epilogue: out = s_in + sigmoid(c) * (act(h) - s_in).
// s_in read from the staged LDS panel; result stored bf16 row-major to global.
// C/D layout: lane l, vgpr v -> M = v + 8*(l>>4), N = l&15 within the tile.
DEVI void epilogue_lds(const __bf16* SinLds, __bf16* Dst, int mbase, int pair, int act,
                       f32x8 accC[4], f32x8 accH[4], int lane) {
    int n = pair * 16 + (lane & 15);
#pragma unroll
    for (int i = 0; i < 4; ++i) {
        int mlocal0 = i * 16 + ((lane >> 4) << 3);
#pragma unroll
        for (int v = 0; v < 8; ++v) {
            int ml = mlocal0 + v;
            float s = (float)SinLds[(size_t)ml * NH + n];
            float g = sigmoidf_(accC[i][v]);
            float a = apply_act(act, accH[i][v]);
            float o = s + g * (a - s);
            Dst[(size_t)(mbase + ml) * NH + n] = (__bf16)o;
        }
    }
}

// ================= kernel 0: init =================
__global__ void darts_init_kernel(char* ws) {
    float*  agg = (float*)(ws + OFF_AGG);
    __bf16* h0  = (__bf16*)(ws + OFF_ST);     // state 0 = h
    int*    bar = (int*)(ws + OFF_BAR);
    size_t tid = (size_t)blockIdx.x * blockDim.x + threadIdx.x;
    size_t stride = (size_t)gridDim.x * blockDim.x;
    if (tid == 0) { bar[0] = 0; bar[1] = 0; }
    for (size_t i = tid; i < ST_ELEMS; i += stride) {
        agg[i] = 0.0f;
        h0[i]  = (__bf16)0.0f;
    }
}

// ================= kernel 1: weight repack fp32 -> bf16 B-fragments =================
DEVI void packElem(__bf16* dst, const float* src, int ld, size_t idx, int K) {
    int frag = (int)(idx >> 9);
    int rem  = (int)(idx & 511);
    int lane = rem >> 4;
    int i    = rem & 15;
    int nKt  = K >> 5;
    int nt   = frag / nKt;
    int kt   = frag - nt * nKt;
    int k    = kt * 32 + ((lane >> 4) << 3) + (i & 7) + ((i >> 3) << 4);   // lanes 0-15: K 0..7,16..23; 16-31: K 8..15,24..31
    int n    = nt * 16 + (lane & 15);
    dst[idx] = (__bf16)src[(size_t)k * ld + n];
}

__global__ void darts_pack_kernel(const float* enc_W, const float* W0, const float* Ws, char* ws) {
    __bf16* pencw = (__bf16*)(ws + OFF_PENCW);
    __bf16* pw0x  = (__bf16*)(ws + OFF_PW0X);
    __bf16* pw0h  = (__bf16*)(ws + OFF_PW0H);
    __bf16* pws   = (__bf16*)(ws + OFF_PWS);
    size_t tid = (size_t)blockIdx.x * blockDim.x + threadIdx.x;
    size_t stride = (size_t)gridDim.x * blockDim.x;
    for (size_t idx = tid; idx < N_PACK_TOTAL; idx += stride) {
        if (idx < N_ENCW) {
            packElem(pencw, enc_W, NE, idx, NE);
        } else if (idx < N_ENCW + N_W0X) {
            packElem(pw0x, W0, N2H, idx - N_ENCW, NE);
        } else if (idx < N_ENCW + N_W0X + N_W0H) {
            packElem(pw0h, W0 + (size_t)NE * N2H, N2H, idx - N_ENCW - N_W0X, NH);
        } else {
            size_t r = idx - N_ENCW - N_W0X - N_W0H;
            int w = (int)(r / N_WSM);
            size_t j = r - (size_t)w * N_WSM;
            packElem(pws + (size_t)w * N_WSM, Ws + (size_t)w * NH * N2H, N2H, j, NH);
        }
    }
}

// ================= kernel 2: embedding gather + encoder GEMM =================
// encoded[s*128+b][e] = (token_embs[words[b][s]] @ enc_W + enc_b) in bf16
__global__ __launch_bounds__(256)
void darts_encode_kernel(const int* words, const float* token_embs, const float* enc_b, char* ws) {
    const __bf16* pencw = (const __bf16*)(ws + OFF_PENCW);
    __bf16* encoded     = (__bf16*)(ws + OFF_ENC);
    int lane = threadIdx.x & 31;
    int wid  = blockIdx.x * 8 + (threadIdx.x >> 5);
    int nw   = gridDim.x * 8;
    const int items = (NS * NB / 16) * (NE / 16);   // 2048 m-tiles * 32 n-tiles
    for (int it = wid; it < items; it += nw) {
        int mt = it >> 5;
        int nt = it & 31;
        int r = mt * 16 + (lane & 15);   // r = s*128 + b
        int b = r & 127;
        int s = r >> 7;
        int word = words[b * NS + s];
        const float* arow = token_embs + (size_t)word * NE;
        f32x8 acc = {};
        for (int kt = 0; kt < NE / 32; ++kt) {
            int k0 = kt * 32 + ((lane >> 4) << 3);
            f32x4 f0 = *(const f32x4*)(arow + k0);
            f32x4 f1 = *(const f32x4*)(arow + k0 + 4);
            f32x4 f2 = *(const f32x4*)(arow + k0 + 16);
            f32x4 f3 = *(const f32x4*)(arow + k0 + 20);
            BF16x16u a;
#pragma unroll
            for (int j = 0; j < 4; ++j) {
                a.v[j]      = (__bf16)f0[j];
                a.v[4 + j]  = (__bf16)f1[j];
                a.v[8 + j]  = (__bf16)f2[j];
                a.v[12 + j] = (__bf16)f3[j];
            }
            bf16x16 bfrag = loadBfrag(pencw, NE / 32, nt, kt, lane);
            acc = wmma_bf16(a.v, bfrag, acc);
        }
        int n = nt * 16 + (lane & 15);
        float bias = enc_b[n];
        int mb = mt * 16 + ((lane >> 4) << 3);
#pragma unroll
        for (int v = 0; v < 8; ++v)
            encoded[(size_t)(mb + v) * NE + n] = (__bf16)(acc[v] + bias);
    }
}

// ================= kernel 3: persistent RNN scan =================
// state indices: 0=h, 1=s0, 2..9 = past1..past8
struct MMd { unsigned char src, w, act, dst; };
__device__ __constant__ MMd LVL_DESCS[8] = {
    {1, 0, 0, 2},                              // level1: l0 tanh    s0   -> p1
    {2, 1, 1, 3}, {2, 2, 2, 4},                // level2: l1 relu p1->p2 ; l2 sigm p1->p3
    {3, 3, 3, 5}, {3, 4, 0, 6}, {4, 5, 1, 7},  // level3: l3 id p2->p4 ; l4 tanh p2->p5 ; l5 relu p3->p6
    {5, 6, 2, 8}, {6, 7, 3, 9}                 // level4: l6 sigm p4->p7 ; l7 id p5->p8
};
__device__ __constant__ int LVL_OFF[4] = {0, 1, 3, 6};
__device__ __constant__ int LVL_CNT[4] = {1, 2, 3, 2};

// Grid-wide sense-reversing barrier. Every thread releases its stores
// (threadfence) before arriving; leader does agent-scope acquire after.
DEVI void grid_barrier(int* bar, int* ssense) {
    __threadfence();
    __syncthreads();
    if (threadIdx.x == 0) {
        int s = *ssense ^ 1;
        int a = __hip_atomic_fetch_add(&bar[0], 1, __ATOMIC_ACQ_REL, __HIP_MEMORY_SCOPE_AGENT);
        if (a == RNN_BLOCKS - 1) {
            __hip_atomic_store(&bar[0], 0, __ATOMIC_RELAXED, __HIP_MEMORY_SCOPE_AGENT);
            __hip_atomic_store(&bar[1], s, __ATOMIC_RELEASE, __HIP_MEMORY_SCOPE_AGENT);
        } else {
            while (__hip_atomic_load(&bar[1], __ATOMIC_ACQUIRE, __HIP_MEMORY_SCOPE_AGENT) != s)
                __builtin_amdgcn_s_sleep(1);
        }
        *ssense = s;
        __threadfence();
    }
    __syncthreads();
}

__global__ __launch_bounds__(RNN_THREADS)
void darts_rnn_kernel(const int* masks, char* ws) {
    int*           bar     = (int*)(ws + OFF_BAR);
    float*         agg     = (float*)(ws + OFF_AGG);
    __bf16*        states  = (__bf16*)(ws + OFF_ST);
    const __bf16*  encoded = (const __bf16*)(ws + OFF_ENC);
    const __bf16*  pw0x    = (const __bf16*)(ws + OFF_PW0X);
    const __bf16*  pw0h    = (const __bf16*)(ws + OFF_PW0H);
    const __bf16*  pws     = (const __bf16*)(ws + OFF_PWS);

    // 64-row x 1024-col bf16 A panel staged per block (128 KB of the 320 KB WGP LDS)
    __shared__ __align__(16) __bf16 Apanel[64 * NH];
    __shared__ int ssense;
    if (threadIdx.x == 0) ssense = 0;
    __syncthreads();

    int tid  = threadIdx.x;
    int lane = threadIdx.x & 31;
    int wv   = threadIdx.x >> 5;                 // wave in block: 0..7
    int pair = (blockIdx.x & 7) * 8 + wv;        // column pair 0..63 (cols 16p / 1024+16p)
    int grp  = blockIdx.x >> 3;                  // group = matmul*2 + mstrip
    int tidg = blockIdx.x * RNN_THREADS + threadIdx.x;
    const int nthreads = RNN_BLOCKS * RNN_THREADS;

    for (int t = 0; t < NS; ++t) {
        // ---- level A: joint = x_t@W0x + h@W0h ; s0 = h + sig(c)*(tanh(hh)-h)
        if (grp < 2) {
            int mbase = (grp & 1) * 64;
            f32x8 accC[4] = {};
            f32x8 accH[4] = {};
            stage_panel_async(encoded + (size_t)t * NB * NE + (size_t)mbase * NE,
                              Apanel, 64 * NE * 2, tid);
            __syncthreads();
            gemm_lds(Apanel, NE, pw0x, NE, pair, accC, accH, lane);
            __syncthreads();
            stage_panel_async(states + (size_t)mbase * NH, Apanel, 64 * NH * 2, tid);
            __syncthreads();
            gemm_lds(Apanel, NH, pw0h, NH, pair, accC, accH, lane);
            epilogue_lds(Apanel, states + ST_ELEMS /*s0*/, mbase, pair, /*tanh*/0,
                         accC, accH, lane);
        }
        grid_barrier(bar, &ssense);

        // ---- levels 1..4 over the DARTS DAG
        for (int lvl = 0; lvl < 4; ++lvl) {
            int cnt = LVL_CNT[lvl];
            if (grp < cnt * 2) {
                MMd d = LVL_DESCS[LVL_OFF[lvl] + (grp >> 1)];
                int mbase = (grp & 1) * 64;
                const __bf16* src = states + (size_t)d.src * ST_ELEMS;
                __bf16*       dst = states + (size_t)d.dst * ST_ELEMS;
                const __bf16* wp  = pws + (size_t)d.w * N_WSM;
                stage_panel_async(src + (size_t)mbase * NH, Apanel, 64 * NH * 2, tid);
                __syncthreads();
                f32x8 accC[4] = {};
                f32x8 accH[4] = {};
                gemm_lds(Apanel, NH, wp, NH, pair, accC, accH, lane);
                epilogue_lds(Apanel, dst, mbase, pair, d.act, accC, accH, lane);
            }
            grid_barrier(bar, &ssense);
        }

        // ---- merge: h_new = mean(past1..8); agg += keep * h_new
        for (int idx = tidg; idx < NB * NH; idx += nthreads) {
            int m = idx >> 10;
            float sum = 0.0f;
#pragma unroll
            for (int s = 2; s <= 9; ++s)
                sum += (float)states[(size_t)s * ST_ELEMS + idx];
            float hnew = sum * 0.125f;
            states[idx] = (__bf16)hnew;                         // h for next step
            float keep = 1.0f - (float)masks[m * NS + t];
            agg[idx] += keep * hnew;
        }
        grid_barrier(bar, &ssense);
    }
}

// ================= kernel 4: decode + outputs =================
// out[0 .. 5375]           = logits  [128][42]
// out[5376 .. 5376+131071] = aggregated [128][1024]
__global__ void darts_decode_kernel(const float* dec_W, const float* dec_b, const char* ws, float* out) {
    const float* agg = (const float*)(ws + OFF_AGG);
    int b = blockIdx.x;
    const float inv = 1.0f / (float)NS;
    for (int n = threadIdx.x; n < NH; n += blockDim.x)
        out[NB * NC + b * NH + n] = agg[(size_t)b * NH + n] * inv;
    for (int c = threadIdx.x; c < NC; c += blockDim.x) {
        float sum = dec_b[c];
        for (int k = 0; k < NH; ++k)
            sum += agg[(size_t)b * NH + k] * inv * dec_W[(size_t)k * NC + c];
        out[b * NC + c] = sum;
    }
}

// ================= host launcher =================
extern "C" void kernel_launch(void* const* d_in, const int* in_sizes, int n_in,
                              void* d_out, int out_size, void* d_ws, size_t ws_size,
                              hipStream_t stream) {
    const int*   words      = (const int*)d_in[0];
    const int*   masks      = (const int*)d_in[1];
    const float* token_embs = (const float*)d_in[2];
    const float* enc_W      = (const float*)d_in[3];
    const float* enc_b      = (const float*)d_in[4];
    const float* W0         = (const float*)d_in[5];
    const float* Ws         = (const float*)d_in[6];
    const float* dec_W      = (const float*)d_in[7];
    const float* dec_b      = (const float*)d_in[8];
    float* out = (float*)d_out;
    char*  ws  = (char*)d_ws;

    darts_init_kernel<<<256, 256, 0, stream>>>(ws);
    darts_pack_kernel<<<4096, 256, 0, stream>>>(enc_W, W0, Ws, ws);
    darts_encode_kernel<<<2048, 256, 0, stream>>>(words, token_embs, enc_b, ws);
    darts_rnn_kernel<<<RNN_BLOCKS, RNN_THREADS, 0, stream>>>(masks, ws);
    darts_decode_kernel<<<NB, 256, 0, stream>>>(dec_W, dec_b, ws, out);
    (void)in_sizes; (void)n_in; (void)out_size; (void)ws_size;
}
